// SurvRankingLoss_3633542332969
// MI455X (gfx1250) — compile-verified, matrix operand order checked by reading, not run on hardware
//
#include <hip/hip_runtime.h>
#include <float.h>

typedef __attribute__((ext_vector_type(16))) _Float16 v16h;
typedef __attribute__((ext_vector_type(8)))  float    v8f;

#define BTOT   8192
#define BLOCK  256
#define ISPLIT 32                // 32 * 256 = 8192 rows
#define JSPLIT 16                // j slices
#define JCHUNK (BTOT / JSPLIT)   // 512 j per block
#define NBLOCKS (ISPLIT * JSPLIT)

// gfx1250 has v_tanh_f32 (one transcendental); fallback uses exp2+rcp.
__device__ __forceinline__ float fast_tanh(float x) {
#if __has_builtin(__builtin_amdgcn_tanhf)
  return __builtin_amdgcn_tanhf(x);
#else
  float e = __builtin_amdgcn_exp2f(x * 2.8853900817779268f);  // exp(2x)
  return 1.0f - 2.0f * __builtin_amdgcn_rcpf(e + 1.0f);
#endif
}

__global__ __launch_bounds__(BLOCK) void surv_pair_kernel(
    const float* __restrict__ z, const float* __restrict__ times,
    const int* __restrict__ cens, float* __restrict__ partial) {
  __shared__ __align__(16) float st[JCHUNK];   // time_j
  __shared__ __align__(16) float sh[JCHUNK];   // 0.5 * z_j
  __shared__ float swn[BLOCK / 32];
  __shared__ float swd[BLOCK / 32];

  const int tid = threadIdx.x;
  const int i   = blockIdx.x * BLOCK + tid;   // row owned by this lane
  const int j0  = blockIdx.y * JCHUNK;

  // Stage the j slice into LDS (SoA; pre-scale z by 0.5).
  for (int k = tid; k < JCHUNK; k += BLOCK) {
    const int j = j0 + k;
    st[k] = times[j];
    sh[k] = 0.5f * z[j];
  }
  __syncthreads();

  // Fold the event mask into the comparison key once per lane:
  // censored rows get ti=+FLT_MAX so (ti < tj) is never true (times are in [0,1)).
  const float ti_raw = times[i];
  const float ti     = (cens[i] == 0) ? ti_raw : FLT_MAX;
  const float hi     = 0.5f * z[i];

  v16h ones;
#pragma unroll
  for (int k = 0; k < 16; ++k) ones[k] = (_Float16)1.0f;

  v8f      cT0 = {};   // two independent WMMA accumulators to break the
  v8f      cT1 = {};   // serial C dependency / hazard stalls
  unsigned cnt = 0;    // wave-uniform masked-pair count (SALU bcnt/add)

  for (int c = 0; c < JCHUNK; c += 32) {
    float tj[32], hj[32];
#pragma unroll
    for (int q = 0; q < 8; ++q) {   // 8x ds_load_b128 per array
      *(float4*)&tj[4 * q] = *(const float4*)&st[c + 4 * q];
      *(float4*)&hj[4 * q] = *(const float4*)&sh[c + 4 * q];
    }

    v16h av0, av1;
#pragma unroll
    for (int k = 0; k < 16; ++k) {
      const bool  lt0 = ti < tj[k];
      const float t0  = fast_tanh(hi - hj[k]);          // tanh((zi-zj)/2)
      av0[k] = (_Float16)(lt0 ? t0 : 0.0f);
      // Same fcmp feeds the select AND the ballot: count is pure SALU
      // (s_bcnt1 + s_add) on the wave-wide compare mask, co-executing free.
      cnt += (unsigned)__builtin_popcount(__builtin_amdgcn_ballot_w32(lt0));

      const bool  lt1 = ti < tj[16 + k];
      const float t1  = fast_tanh(hi - hj[16 + k]);
      av1[k] = (_Float16)(lt1 ? t1 : 0.0f);
      cnt += (unsigned)__builtin_popcount(__builtin_amdgcn_ballot_w32(lt1));
    }
    // D = A x ones + C : two independent 512-way f32 adders on the matrix unit.
    cT0 = __builtin_amdgcn_wmma_f32_16x16x32_f16(
        false, av0, false, ones, (short)0, cT0, false, false);
    cT1 = __builtin_amdgcn_wmma_f32_16x16x32_f16(
        false, av1, false, ones, (short)0, cT1, false, false);
  }

  // Per-lane fold of the accumulator VGPRs.
  float pn = 0.0f;
#pragma unroll
  for (int k = 0; k < 8; ++k) pn += cT0[k] + cT1[k];

  // wave32 reduction (deterministic); cnt is already wave-uniform.
  for (int off = 16; off; off >>= 1) pn += __shfl_xor(pn, off, 32);

  const int wid = tid >> 5;
  if ((tid & 31) == 0) {
    swn[wid] = pn;
    swd[wid] = (float)cnt;   // wave count (covers all 32 lanes of this wave)
  }
  __syncthreads();
  if (tid == 0) {
    float bn = 0.0f, bd = 0.0f;
#pragma unroll
    for (int w = 0; w < BLOCK / 32; ++w) { bn += swn[w]; bd += swd[w]; }
    const int bid = blockIdx.y * gridDim.x + blockIdx.x;
    partial[2 * bid]     = bn;   // 16x-replicated sum of masked tanh
    partial[2 * bid + 1] = bd;   // exact masked-pair count
  }
}

__global__ __launch_bounds__(256) void surv_finalize_kernel(
    const float* __restrict__ partial, float* __restrict__ out) {
  __shared__ float sn[8];
  __shared__ float sd[8];
  const int t = threadIdx.x;
  float pn = 0.0f, pd = 0.0f;
  for (int k = t; k < NBLOCKS; k += 256) {
    pn += partial[2 * k];
    pd += partial[2 * k + 1];
  }
  for (int off = 16; off; off >>= 1) {
    pn += __shfl_xor(pn, off, 32);
    pd += __shfl_xor(pd, off, 32);
  }
  if ((t & 31) == 0) { sn[t >> 5] = pn; sd[t >> 5] = pd; }
  __syncthreads();
  if (t == 0) {
    float T = 0.0f, M = 0.0f;
#pragma unroll
    for (int w = 0; w < 8; ++w) { T += sn[w]; M += sd[w]; }
    // sum(mask*sigmoid) = 0.5*(sum(mask*tanh) + sum(mask));  T carries a 16x
    // column-replication factor from the ones-matrix WMMA accumulation.
    const float num = 0.5f * (T * (1.0f / 16.0f) + M);
    out[0] = -num / M;
  }
}

extern "C" void kernel_launch(void* const* d_in, const int* in_sizes, int n_in,
                              void* d_out, int out_size, void* d_ws, size_t ws_size,
                              hipStream_t stream) {
  const float* z     = (const float*)d_in[0];
  const float* times = (const float*)d_in[1];
  const int*   cens  = (const int*)d_in[2];
  float* partial = (float*)d_ws;            // 2 * NBLOCKS floats = 4 KB
  float* out     = (float*)d_out;

  dim3 grid(ISPLIT, JSPLIT);
  surv_pair_kernel<<<grid, dim3(BLOCK), 0, stream>>>(z, times, cens, partial);
  surv_finalize_kernel<<<dim3(1), dim3(256), 0, stream>>>(partial, out);
}
